// RelativeGlobalMultiheadAttention_30588757082711
// MI455X (gfx1250) — matrix operand order repaired and checked by reading, not compiled, not run on hardware
//
#include <hip/hip_runtime.h>
#include <hip/hip_bf16.h>

// ---------------------------------------------------------------------------
// Relative global multi-head attention, CDNA5 (gfx1250), wave32 WMMA bf16.
// B=4, S=1024, D=1024, H=16, dh=64.
// ---------------------------------------------------------------------------

#define B_  4
#define S_  1024
#define D_  1024
#define H_  16
#define DH_ 64

typedef __attribute__((ext_vector_type(16))) __bf16 v16bf;
typedef __attribute__((ext_vector_type(8)))  float  v8f;

#define WMMA_BF16(A, Bm, C) \
  __builtin_amdgcn_wmma_f32_16x16x32_bf16(false, (A), false, (Bm), (short)0, (C), false, false)

static __device__ __forceinline__ unsigned short f32_to_bf16_raw(float f) {
  unsigned int u = __builtin_bit_cast(unsigned int, f);
  unsigned int r = u + 0x7fffu + ((u >> 16) & 1u);  // round-to-nearest-even
  return (unsigned short)(r >> 16);
}

// A-fragment (16x32 MxK, bf16): lane holds row M = lane%16.
// ISA layout: VGPR v<4 -> K = {2v,2v+1} + 8*half ; v>=4 -> K = 16+{2(v-4),...} + 8*half.
// Per lane this is two contiguous 16-byte runs -> two b128 loads.
static __device__ __forceinline__ v16bf load_frag_a(const unsigned short* rowptr, int half) {
  union { v16bf bf; uint4 q[2]; } f;
  f.q[0] = *(const uint4*)(rowptr + 8 * half);
  f.q[1] = *(const uint4*)(rowptr + 16 + 8 * half);
  return f.bf;
}

// B-fragment (32x16 KxN, bf16): lane holds column N = lane%16; same K order as A:
// elements 0-7 -> K = 8*half + 0..7, elements 8-15 -> K = 16 + 8*half + 0..7.
// rowptr = this lane's column data, 32 contiguous bf16 over K -> two b128 loads.
static __device__ __forceinline__ v16bf load_frag_b(const unsigned short* rowptr, int half) {
  union { v16bf bf; uint4 q[2]; } f;
  f.q[0] = *(const uint4*)(rowptr + 8 * half);
  f.q[1] = *(const uint4*)(rowptr + 16 + 8 * half);
  return f.bf;
}

static __device__ __forceinline__ v16bf zero_frag() {
  union { v16bf bf; uint4 q[2]; } f;
  f.q[0] = make_uint4(0u, 0u, 0u, 0u);
  f.q[1] = make_uint4(0u, 0u, 0u, 0u);
  return f.bf;
}

// ---------------------------------------------------------------------------
// fp32 -> bf16 conversion (grid-stride)
// ---------------------------------------------------------------------------
__global__ void cvt_f32_to_bf16(const float* __restrict__ src,
                                unsigned short* __restrict__ dst, int n) {
  int i = blockIdx.x * blockDim.x + threadIdx.x;
  int stride = gridDim.x * blockDim.x;
  for (; i < n; i += stride) dst[i] = f32_to_bf16_raw(src[i]);
}

// ---------------------------------------------------------------------------
// Projection GEMM: Y = X(bf16) @ W^T + bias, written bf16 head-major
// [B*H][S][dh].  Block tile 128x128, 8 waves (4x2), wave tile 32x64
// (2x4 WMMA tiles, A-fragments reused across 4 B-fragments).
// blockIdx.z selects Q / K / V.
// ---------------------------------------------------------------------------
__global__ __launch_bounds__(256) void proj_qkv_wmma(
    const unsigned short* __restrict__ xb,
    const unsigned short* __restrict__ wqb,
    const unsigned short* __restrict__ wkb,
    const unsigned short* __restrict__ wvb,
    const float* __restrict__ bq, const float* __restrict__ bk,
    const float* __restrict__ bv,
    unsigned short* __restrict__ qb, unsigned short* __restrict__ kb,
    unsigned short* __restrict__ vb) {
  const unsigned short* W;
  const float* bias;
  unsigned short* out;
  if (blockIdx.z == 0)      { W = wqb; bias = bq; out = qb; }
  else if (blockIdx.z == 1) { W = wkb; bias = bk; out = kb; }
  else                      { W = wvb; bias = bv; out = vb; }

  __shared__ unsigned short Asl[128][32];  // x tile, row-major (M x K)
  __shared__ unsigned short Bsl[128][32];  // W tile, N-major rows (N x K)

  const int tid  = threadIdx.x;
  const int lane = tid & 31, wid = tid >> 5;
  const int ln   = lane & 15, half = lane >> 4;
  const int wm   = wid & 3,  wn  = wid >> 2;  // 4 M-waves x 2 N-waves
  const int m0   = blockIdx.x * 128;
  const int n0   = blockIdx.y * 128;

  v8f zero = {};
  v8f acc[2][4];
#pragma unroll
  for (int mi = 0; mi < 2; ++mi)
#pragma unroll
    for (int ni = 0; ni < 4; ++ni) acc[mi][ni] = zero;

  for (int k0 = 0; k0 < D_; k0 += 32) {
#pragma unroll
    for (int e = tid; e < 128 * 32; e += 256) {
      int r = e >> 5, c = e & 31;
      Asl[r][c] = xb[(size_t)(m0 + r) * D_ + (k0 + c)];
    }
#pragma unroll
    for (int e = tid; e < 128 * 32; e += 256) {
      int r = e >> 5, c = e & 31;
      Bsl[r][c] = W[(size_t)(n0 + r) * D_ + (k0 + c)];
    }
    // Prefetch next K-slab while this one is consumed (global_prefetch_b8).
    if (k0 + 32 < D_) {
      __builtin_prefetch(xb + (size_t)(m0 + (tid & 127)) * D_ + (k0 + 32), 0, 0);
      __builtin_prefetch(W  + (size_t)(n0 + (tid & 127)) * D_ + (k0 + 32), 0, 0);
    }
    __syncthreads();

    v16bf af0 = load_frag_a(&Asl[wm * 32 + ln][0], half);
    v16bf af1 = load_frag_a(&Asl[wm * 32 + 16 + ln][0], half);
#pragma unroll
    for (int ni = 0; ni < 4; ++ni) {
      v16bf bfn = load_frag_b(&Bsl[wn * 64 + ni * 16 + ln][0], half);
      acc[0][ni] = WMMA_BF16(af0, bfn, acc[0][ni]);
      acc[1][ni] = WMMA_BF16(af1, bfn, acc[1][ni]);
    }
    __syncthreads();
  }

  // Epilogue: + bias, convert to bf16, scatter to head-major [B*H][S][dh].
#pragma unroll
  for (int mi = 0; mi < 2; ++mi)
#pragma unroll
    for (int ni = 0; ni < 4; ++ni)
#pragma unroll
      for (int v = 0; v < 8; ++v) {
        int gm = m0 + wm * 32 + mi * 16 + v + 8 * half;  // global row in B*S
        int gn = n0 + wn * 64 + ni * 16 + ln;            // global col in D
        float y = acc[mi][ni][v] + bias[gn];
        int b = gm >> 10, s = gm & (S_ - 1);
        int h = gn >> 6,  d = gn & (DH_ - 1);
        out[(((size_t)(b * H_ + h)) * S_ + s) * DH_ + d] = f32_to_bf16_raw(y);
      }
}

// ---------------------------------------------------------------------------
// Flash attention with relative (skewed) positional scores.
// Block: 128 threads = 4 waves; 64 query rows of one (b,h); wave = 16 rows.
// Key loop in steps of 32; scores = QK^T + Srel band (Er WMMAs + LDS diagonal
// gather); online softmax; O += P@V.
// ---------------------------------------------------------------------------
__global__ __launch_bounds__(128) void rel_attn_wmma(
    const unsigned short* __restrict__ qb,
    const unsigned short* __restrict__ kb,
    const unsigned short* __restrict__ vb,
    const unsigned short* __restrict__ erb,
    float* __restrict__ out) {
  __shared__ unsigned short Ksl[32][64];    // K tile: rows j, cols d
  __shared__ unsigned short Vtsl[64][32];   // V^T tile: rows d, cols j
  __shared__ float          P48[4][16][48]; // per-wave q@Er^T band
  __shared__ unsigned short Pw[4][16][32];  // per-wave bf16 P transpose buffer

  const int tid  = threadIdx.x;
  const int lane = tid & 31, wid = tid >> 5;
  const int ln   = lane & 15, half = lane >> 4;
  const int r0   = blockIdx.x * 64;
  const int wr0  = r0 + wid * 16;           // wave's 16 query rows start
  const int bh   = blockIdx.y;
  const int b    = bh >> 4, h = bh & 15;

  // Q fragments for this wave's 16 rows (dh=64 -> two K=32 chunks)
  const unsigned short* qrow = qb + ((size_t)bh * S_ + wr0 + ln) * DH_;
  v16bf qf[2];
  qf[0] = load_frag_a(qrow, half);
  qf[1] = load_frag_a(qrow + 32, half);

  v8f zero = {};
  v8f o[4];
#pragma unroll
  for (int nt = 0; nt < 4; ++nt) o[nt] = zero;
  float mrow[8], lrow[8];
#pragma unroll
  for (int v = 0; v < 8; ++v) { mrow[v] = -3.0e38f; lrow[v] = 0.0f; }

  const float scale = 0.125f;  // 1/sqrt(dh)

  for (int j0 = 0; j0 < r0 + 64; j0 += 32) {
    // Stage 32 keys and values (V transposed) into LDS.
#pragma unroll
    for (int e = tid; e < 32 * 64; e += 128) {
      int j = e >> 6, d = e & 63;
      Ksl[j][d] = kb[((size_t)bh * S_ + j0 + j) * DH_ + d];
    }
#pragma unroll
    for (int e = tid; e < 32 * 64; e += 128) {
      int d = e >> 5, j = e & 31;
      Vtsl[d][j] = vb[((size_t)bh * S_ + j0 + j) * DH_ + d];
    }
    __syncthreads();

    if (j0 <= wr0 + 15) {  // wave-uniform: this key block intersects causal region
      // ---- QK^T: 16x32 scores (two 16-col subtiles, two K chunks) ----
      v8f s0 = zero, s1 = zero;
#pragma unroll
      for (int kc = 0; kc < 2; ++kc) {
        v16bf bk0 = load_frag_b(&Ksl[ln][kc * 32], half);
        v16bf bk1 = load_frag_b(&Ksl[16 + ln][kc * 32], half);
        s0 = WMMA_BF16(qf[kc], bk0, s0);
        s1 = WMMA_BF16(qf[kc], bk1, s1);
      }

      // ---- Srel band: P48[m][c] = q[wr0+m] . Er[rstart+c], c in [0,48) ----
      // Srel[i][j] = (q @ Er^T)[i][j - i + S - 1]; band start rstart = j0-wr0+S-16.
      const int rstart = j0 - wr0 + (S_ - 16);
#pragma unroll
      for (int t = 0; t < 3; ++t) {
        v8f pc = zero;
        int r = rstart + t * 16 + ln;
#pragma unroll
        for (int kc = 0; kc < 2; ++kc) {
          v16bf be = (r < S_) ? load_frag_b(erb + (size_t)r * DH_ + kc * 32, half)
                              : zero_frag();
          pc = WMMA_BF16(qf[kc], be, pc);
        }
#pragma unroll
        for (int v = 0; v < 8; ++v)
          P48[wid][v + 8 * half][t * 16 + ln] = pc[v];
      }
      // Wave-local LDS: DS ops from the same wave are in-order; no barrier needed.

      // ---- add Srel diagonal, causal mask, online softmax ----
#pragma unroll
      for (int v = 0; v < 8; ++v) {
        int m = v + 8 * half;
        int i = wr0 + m;
        int jj0 = ln, jj1 = 16 + ln;
        float sc0 = (s0[v] + P48[wid][m][15 + jj0 - m]) * scale;
        float sc1 = (s1[v] + P48[wid][m][15 + jj1 - m]) * scale;
        if (j0 + jj0 > i) sc0 = -1.0e6f;
        if (j0 + jj1 > i) sc1 = -1.0e6f;

        float tmax = fmaxf(sc0, sc1);
#pragma unroll
        for (int off = 1; off < 16; off <<= 1)
          tmax = fmaxf(tmax, __shfl_xor(tmax, off));
        float mnew  = fmaxf(mrow[v], tmax);
        float alpha = __expf(mrow[v] - mnew);
        mrow[v] = mnew;

        float p0 = __expf(sc0 - mnew);
        float p1 = __expf(sc1 - mnew);
        float rs = p0 + p1;
#pragma unroll
        for (int off = 1; off < 16; off <<= 1) rs += __shfl_xor(rs, off);
        lrow[v] = lrow[v] * alpha + rs;

#pragma unroll
        for (int nt = 0; nt < 4; ++nt) o[nt][v] *= alpha;

        Pw[wid][m][jj0] = f32_to_bf16_raw(p0);
        Pw[wid][m][jj1] = f32_to_bf16_raw(p1);
      }

      // ---- O += P(16x32) @ V(32x64) ----
      v16bf pa = load_frag_a(&Pw[wid][ln][0], half);
#pragma unroll
      for (int nt = 0; nt < 4; ++nt) {
        v16bf bvf = load_frag_b(&Vtsl[nt * 16 + ln][0], half);
        o[nt] = WMMA_BF16(pa, bvf, o[nt]);
      }
    }
    __syncthreads();
  }

  // ---- finalize: divide by row sums, write [B,S,H,dh] fp32 ----
#pragma unroll
  for (int v = 0; v < 8; ++v) {
    int m = v + 8 * half;
    int i = wr0 + m;
    float inv = 1.0f / lrow[v];
#pragma unroll
    for (int nt = 0; nt < 4; ++nt) {
      int d = nt * 16 + ln;
      out[(((size_t)b * S_ + i) * H_ + h) * DH_ + d] = o[nt][v] * inv;
    }
  }
}

// ---------------------------------------------------------------------------
// Host launcher
// ---------------------------------------------------------------------------
extern "C" void kernel_launch(void* const* d_in, const int* in_sizes, int n_in,
                              void* d_out, int out_size, void* d_ws,
                              size_t ws_size, hipStream_t stream) {
  (void)in_sizes; (void)n_in; (void)out_size; (void)ws_size;
  const float* x  = (const float*)d_in[0];
  const float* Wq = (const float*)d_in[1];
  const float* bq = (const float*)d_in[2];
  const float* Wk = (const float*)d_in[3];
  const float* bk = (const float*)d_in[4];
  const float* Wv = (const float*)d_in[5];
  const float* bv = (const float*)d_in[6];
  const float* Er = (const float*)d_in[7];
  float* out = (float*)d_out;

  char* ws = (char*)d_ws;
  size_t off = 0;
  auto alloc = [&](size_t bytes) -> char* {
    char* p = ws + off;
    off += bytes;
    off = (off + 255) & ~(size_t)255;
    return p;
  };
  unsigned short* xb  = (unsigned short*)alloc((size_t)B_ * S_ * D_ * 2);
  unsigned short* wqb = (unsigned short*)alloc((size_t)D_ * D_ * 2);
  unsigned short* wkb = (unsigned short*)alloc((size_t)D_ * D_ * 2);
  unsigned short* wvb = (unsigned short*)alloc((size_t)D_ * D_ * 2);
  unsigned short* erb = (unsigned short*)alloc((size_t)S_ * DH_ * 2);
  unsigned short* qb  = (unsigned short*)alloc((size_t)B_ * H_ * S_ * DH_ * 2);
  unsigned short* kb  = (unsigned short*)alloc((size_t)B_ * H_ * S_ * DH_ * 2);
  unsigned short* vb  = (unsigned short*)alloc((size_t)B_ * H_ * S_ * DH_ * 2);

  // 1) fp32 -> bf16 conversions
  cvt_f32_to_bf16<<<1024, 256, 0, stream>>>(x,  xb,  B_ * S_ * D_);
  cvt_f32_to_bf16<<<1024, 256, 0, stream>>>(Wq, wqb, D_ * D_);
  cvt_f32_to_bf16<<<1024, 256, 0, stream>>>(Wk, wkb, D_ * D_);
  cvt_f32_to_bf16<<<1024, 256, 0, stream>>>(Wv, wvb, D_ * D_);
  cvt_f32_to_bf16<<<256,  256, 0, stream>>>(Er, erb, S_ * DH_);

  // 2) QKV projections (z selects Q/K/V), block tile 128x128
  dim3 gproj(B_ * S_ / 128, D_ / 128, 3);
  proj_qkv_wmma<<<gproj, 256, 0, stream>>>(xb, wqb, wkb, wvb, bq, bk, bv,
                                           qb, kb, vb);

  // 3) causal attention with relative positions
  dim3 gattn(S_ / 64, B_ * H_);
  rel_attn_wmma<<<gattn, 128, 0, stream>>>(qb, kb, vb, erb, out);
}
